// S4DLayer_22153441313128
// MI455X (gfx1250) — compile-verified, matrix operand order checked by reading, not run on hardware
//
#include <hip/hip_runtime.h>
#include <hip/hip_bf16.h>
#include <math.h>

// ---------------------------------------------------------------------------
// S4D layer on gfx1250: chunked-scan reformulation of the FFT convolution.
//   N = 64 complex states, L = 262144, chunk B = 256, M = L/B = 1024 chunks.
// Pipeline:
//   setup -> u->f16 -> state GEMM (WMMA) -> 2-level scan -> fused out GEMM (WMMA)
// ---------------------------------------------------------------------------

typedef __attribute__((ext_vector_type(16))) _Float16 v16h;
typedef __attribute__((ext_vector_type(8)))  float    v8f;

#define S4D_N    64
#define S4D_L    262144
#define S4D_B    256
#define S4D_M    (S4D_L / S4D_B)   // 1024 chunks
#define S4D_2N   128
#define S4D_G    32                // chunks per scan group
#define S4D_NG   (S4D_M / S4D_G)   // 32 groups

// Workspace byte offsets
#define OFF_ZB  0u                                    // 128 floats (z^B re|im)
#define OFF_V   (OFF_ZB + 128u * 4u)                  // 128x256 f16
#define OFF_W   (OFF_V + 128u * 256u * 2u)            // 256x128 f16
#define OFF_T   (OFF_W + 256u * 128u * 2u)            // 256x256 f16
#define OFF_UH  (OFF_T + 256u * 256u * 2u)            // L f16
#define OFF_S   (OFF_UH + (unsigned)S4D_L * 2u)       // M x 128 f32 (chunk-major)
#define OFF_XH  (OFF_S + (unsigned)S4D_M * 128u * 4u) // M x 128 f16 (chunk-major)
#define OFF_GF  (OFF_XH + (unsigned)S4D_M * 128u * 2u)// NG x 128 f32 group finals
#define OFF_CX  (OFF_GF + (unsigned)S4D_NG * 128u * 4u)// NG x 128 f32 excl carries

// K-offset within a 16x32 f16 A/B fragment for half-index i, lane-half h
__device__ __forceinline__ int s4d_koff(int i, int h) {
    return (i < 8) ? (8 * h + i) : (16 + 8 * h + (i - 8));
}

// Load 16x32 f16 A fragment from row-major matrix (leading dim ld) at (row0,k0)
__device__ __forceinline__ v16h s4d_load_a(const _Float16* A, int ld, int row0,
                                           int k0, int lane) {
    int m = lane & 15, h = lane >> 4;
    v16h a;
#pragma unroll
    for (int i = 0; i < 16; ++i) a[i] = A[(row0 + m) * ld + k0 + s4d_koff(i, h)];
    return a;
}

// ---------------------------------------------------------------------------
// Kernel 1: setup.  1 block x 256 threads.
// Builds z^B, V (state Vandermonde), W (cross-output), T (local Toeplitz + D).
// ---------------------------------------------------------------------------
__global__ void s4d_setup(const float* __restrict__ A_re,
                          const float* __restrict__ A_im,
                          const float* __restrict__ C,
                          const float* __restrict__ D,
                          const float* __restrict__ log_step,
                          float* __restrict__ zB,
                          _Float16* __restrict__ V,
                          _Float16* __restrict__ W,
                          _Float16* __restrict__ T) {
    __shared__ float s_cr[S4D_N], s_ci[S4D_N];        // coeff = Ct*(z-1)/A
    __shared__ float s_dr[S4D_N], s_di[S4D_N];        // dt*A
    __shared__ float s_K[S4D_B];                      // kernel head K[0..B-1]
    const int tid = threadIdx.x;
    const float dt = expf(log_step[0]);

    if (tid < S4D_N) {
        float ar = A_re[tid], ai = A_im[tid];
        float dr = dt * ar, di = dt * ai;
        s_dr[tid] = dr; s_di[tid] = di;
        float er = expf(dr), sn, cn;
        sincosf(di, &sn, &cn);
        float zr = er * cn, zi = er * sn;             // z = exp(dt*A)
        float nr = zr - 1.f, ni = zi;                 // z - 1
        float inv = 1.f / (ar * ar + ai * ai);
        float qr = (nr * ar + ni * ai) * inv;         // (z-1)/A
        float qi = (ni * ar - nr * ai) * inv;
        float cr = C[2 * tid + 0], ci = C[2 * tid + 1];
        s_cr[tid] = cr * qr - ci * qi;
        s_ci[tid] = cr * qi + ci * qr;
        // z^B = exp(B * dt*A), phase computed once for accuracy
        float erB = expf((float)S4D_B * dr), sB, cB;
        sincosf((float)S4D_B * di, &sB, &cB);
        zB[tid]          = erB * cB;
        zB[S4D_N + tid]  = erB * sB;
    }
    __syncthreads();

    // K[l] = Re( sum_n coeff_n * z_n^l ), l = tid in [0,256)
    {
        float acc = 0.f;
        for (int n = 0; n < S4D_N; ++n) {
            float er = expf((float)tid * s_dr[n]), sn, cn;
            sincosf((float)tid * s_di[n], &sn, &cn);
            acc += s_cr[n] * er * cn - s_ci[n] * er * sn;
        }
        s_K[tid] = acc;
    }
    __syncthreads();

    // T[t][s] = K[t-s] (s<t), K[0]+D (s==t), 0 (s>t)   -- row t = tid
    {
        const float Dv = D[0];
        for (int s = 0; s < S4D_B; ++s) {
            float v = (s > tid) ? 0.f : ((s == tid) ? (s_K[0] + Dv) : s_K[tid - s]);
            T[tid * S4D_B + s] = (_Float16)v;
        }
    }
    // V[n][s] = Re(z_n^{B-1-s}),  V[64+n][s] = Im(...)   -- column s = tid
    {
        float p = (float)(S4D_B - 1 - tid);
        for (int n = 0; n < S4D_N; ++n) {
            float er = expf(p * s_dr[n]), sn, cn;
            sincosf(p * s_di[n], &sn, &cn);
            V[n * S4D_B + tid]            = (_Float16)(er * cn);
            V[(S4D_N + n) * S4D_B + tid]  = (_Float16)(er * sn);
        }
    }
    // W[t][n] = Re(coeff_n z_n^{t+1}),  W[t][64+n] = -Im(...)   -- row t = tid
    {
        float p = (float)(tid + 1);
        for (int n = 0; n < S4D_N; ++n) {
            float er = expf(p * s_dr[n]), sn, cn;
            sincosf(p * s_di[n], &sn, &cn);
            float zr = er * cn, zi = er * sn;
            W[tid * S4D_2N + n]          = (_Float16)( s_cr[n] * zr - s_ci[n] * zi);
            W[tid * S4D_2N + S4D_N + n]  = (_Float16)(-(s_cr[n] * zi + s_ci[n] * zr));
        }
    }
}

// ---------------------------------------------------------------------------
// Kernel 2: convert u -> f16
// ---------------------------------------------------------------------------
__global__ void s4d_u2h(const float* __restrict__ u, _Float16* __restrict__ uh) {
    int i = blockIdx.x * blockDim.x + threadIdx.x;
    uh[i] = (_Float16)u[i];
}

// ---------------------------------------------------------------------------
// Kernel 3: state GEMM.  S(128 x 1024) = V(128x256) @ U(256x1024)
// U[s][j] = u[j*256+s].  One wave (32 threads) per 16x16 output tile.
// Grid: 8 row-tiles x 64 col-tiles = 512 blocks.  S stored chunk-major.
// ---------------------------------------------------------------------------
__global__ void s4d_state_gemm(const _Float16* __restrict__ V,
                               const _Float16* __restrict__ uh,
                               float* __restrict__ S) {
    const int tile = blockIdx.x;
    const int r0 = (tile & 7) * 16;          // state-row tile
    const int c0 = (tile >> 3) * 16;         // chunk-col tile
    const int lane = threadIdx.x;
    const int m = lane & 15, h = lane >> 4;
    v8f acc = {};
#pragma unroll
    for (int k0 = 0; k0 < S4D_B; k0 += 32) {
        v16h a = s4d_load_a(V, S4D_B, r0, k0, lane);
        v16h b;
#pragma unroll
        for (int i = 0; i < 16; ++i)
            b[i] = uh[(c0 + m) * S4D_B + k0 + s4d_koff(i, h)];
        acc = __builtin_amdgcn_wmma_f32_16x16x32_f16(false, a, false, b,
                                                     (short)0, acc, false, false);
    }
#pragma unroll
    for (int v = 0; v < 8; ++v)
        S[(c0 + m) * S4D_2N + r0 + v + 8 * h] = acc[v];
}

// ---------------------------------------------------------------------------
// Kernel 4a: local scans.  Group g (32 chunks), state n: in-place prefix over S.
// Grid: 32 blocks x 64 threads.  Group-final states -> Gf.
// ---------------------------------------------------------------------------
__global__ void s4d_scan_local(const float* __restrict__ zB,
                               float* __restrict__ S,
                               float* __restrict__ Gf) {
    const int n = threadIdx.x;               // state
    const int g = blockIdx.x;                // group
    const float br = zB[n], bi = zB[S4D_N + n];
    float xr = 0.f, xi = 0.f;
    for (int jl = 0; jl < S4D_G; ++jl) {
        const int j = g * S4D_G + jl;
        float sr = S[j * S4D_2N + n];
        float si = S[j * S4D_2N + S4D_N + n];
        float nxr = br * xr - bi * xi + sr;
        float nxi = br * xi + bi * xr + si;
        xr = nxr; xi = nxi;
        S[j * S4D_2N + n]          = xr;     // local prefix, in place
        S[j * S4D_2N + S4D_N + n]  = xi;
    }
    Gf[g * S4D_2N + n]          = xr;
    Gf[g * S4D_2N + S4D_N + n]  = xi;
}

// ---------------------------------------------------------------------------
// Kernel 4b: scan over 32 group aggregates.  1 block x 64 threads.
// Cex[g] = state at end of group g-1 (exclusive carry); decay z^(B*G) by
// 5 complex squarings of z^B.
// ---------------------------------------------------------------------------
__global__ void s4d_scan_groups(const float* __restrict__ zB,
                                const float* __restrict__ Gf,
                                float* __restrict__ Cex) {
    const int n = threadIdx.x;
    float wr = zB[n], wi = zB[S4D_N + n];    // z^B
#pragma unroll
    for (int t = 0; t < 5; ++t) {            // -> z^(B*32)
        float nr = wr * wr - wi * wi;
        float ni = 2.f * wr * wi;
        wr = nr; wi = ni;
    }
    float cr = 0.f, ci = 0.f;
    for (int g = 0; g < S4D_NG; ++g) {
        Cex[g * S4D_2N + n]          = cr;
        Cex[g * S4D_2N + S4D_N + n]  = ci;
        float gr = Gf[g * S4D_2N + n];
        float gi = Gf[g * S4D_2N + S4D_N + n];
        float nr = wr * cr - wi * ci + gr;
        float ni = wr * ci + wi * cr + gi;
        cr = nr; ci = ni;
    }
}

// ---------------------------------------------------------------------------
// Kernel 4c: apply carries.  X[j] = local[j] + z^(B*(jl+1)) * Cex[g]; emit f16.
// Grid: 32 blocks x 64 threads.
// ---------------------------------------------------------------------------
__global__ void s4d_scan_apply(const float* __restrict__ zB,
                               const float* __restrict__ S,
                               const float* __restrict__ Cex,
                               _Float16* __restrict__ Xh) {
    const int n = threadIdx.x;
    const int g = blockIdx.x;
    const float br = zB[n], bi = zB[S4D_N + n];
    const float cr = Cex[g * S4D_2N + n];
    const float ci = Cex[g * S4D_2N + S4D_N + n];
    float pr = br, pi = bi;                  // z^(B*(jl+1)), starts at z^B
    for (int jl = 0; jl < S4D_G; ++jl) {
        const int j = g * S4D_G + jl;
        float xr = S[j * S4D_2N + n]          + (pr * cr - pi * ci);
        float xi = S[j * S4D_2N + S4D_N + n]  + (pr * ci + pi * cr);
        Xh[j * S4D_2N + n]          = (_Float16)xr;
        Xh[j * S4D_2N + S4D_N + n]  = (_Float16)xi;
        float nr = pr * br - pi * bi;
        float ni = pr * bi + pi * br;
        pr = nr; pi = ni;
    }
}

// ---------------------------------------------------------------------------
// Kernel 5: output GEMM (fused).
//   Y(256x1024) = T(256x256) @ U(256x1024)  +  W(256x128) @ Xshift(128x1024)
//   Xshift[:,j] = X[:,j-1] (zero for j==0).   y[j*256+t] = Y[t][j].
// One wave per 16x16 tile; 16 x 64 = 1024 blocks.
// ---------------------------------------------------------------------------
__global__ void s4d_out_gemm(const _Float16* __restrict__ T,
                             const _Float16* __restrict__ W,
                             const _Float16* __restrict__ uh,
                             const _Float16* __restrict__ Xh,
                             float* __restrict__ out) {
    const int tile = blockIdx.x;
    const int t0 = (tile & 15) * 16;         // within-chunk time tile
    const int j0 = (tile >> 4) * 16;         // chunk tile
    const int lane = threadIdx.x;
    const int m = lane & 15, h = lane >> 4;
    v8f acc = {};
    // local lower-triangular Toeplitz part (K = 256)
#pragma unroll
    for (int k0 = 0; k0 < S4D_B; k0 += 32) {
        v16h a = s4d_load_a(T, S4D_B, t0, k0, lane);
        v16h b;
#pragma unroll
        for (int i = 0; i < 16; ++i)
            b[i] = uh[(j0 + m) * S4D_B + k0 + s4d_koff(i, h)];
        acc = __builtin_amdgcn_wmma_f32_16x16x32_f16(false, a, false, b,
                                                     (short)0, acc, false, false);
    }
    // cross-chunk carry part (K = 128), B fragment from X[:, j-1]
    const int jb = j0 + m;
    const int jsrc = (jb > 0) ? (jb - 1) : 0;
#pragma unroll
    for (int k0 = 0; k0 < S4D_2N; k0 += 32) {
        v16h a = s4d_load_a(W, S4D_2N, t0, k0, lane);
        v16h b;
#pragma unroll
        for (int i = 0; i < 16; ++i) {
            _Float16 v = Xh[jsrc * S4D_2N + k0 + s4d_koff(i, h)];
            b[i] = (jb > 0) ? v : (_Float16)0.f;
        }
        acc = __builtin_amdgcn_wmma_f32_16x16x32_f16(false, a, false, b,
                                                     (short)0, acc, false, false);
    }
#pragma unroll
    for (int v = 0; v < 8; ++v)
        out[(j0 + m) * S4D_B + t0 + v + 8 * h] = acc[v];
}

// ---------------------------------------------------------------------------
extern "C" void kernel_launch(void* const* d_in, const int* in_sizes, int n_in,
                              void* d_out, int out_size, void* d_ws, size_t ws_size,
                              hipStream_t stream) {
    (void)in_sizes; (void)n_in; (void)out_size; (void)ws_size;
    const float* u        = (const float*)d_in[0];
    const float* A_re     = (const float*)d_in[1];
    const float* A_im     = (const float*)d_in[2];
    const float* C        = (const float*)d_in[3];
    const float* D        = (const float*)d_in[4];
    const float* log_step = (const float*)d_in[5];
    float* out = (float*)d_out;

    char* ws = (char*)d_ws;
    float*    zB  = (float*)(ws + OFF_ZB);
    _Float16* V   = (_Float16*)(ws + OFF_V);
    _Float16* W   = (_Float16*)(ws + OFF_W);
    _Float16* T   = (_Float16*)(ws + OFF_T);
    _Float16* uh  = (_Float16*)(ws + OFF_UH);
    float*    S   = (float*)(ws + OFF_S);
    _Float16* Xh  = (_Float16*)(ws + OFF_XH);
    float*    Gf  = (float*)(ws + OFF_GF);
    float*    Cex = (float*)(ws + OFF_CX);

    s4d_setup<<<1, 256, 0, stream>>>(A_re, A_im, C, D, log_step, zB, V, W, T);
    s4d_u2h<<<S4D_L / 256, 256, 0, stream>>>(u, uh);
    s4d_state_gemm<<<(S4D_2N / 16) * (S4D_M / 16), 32, 0, stream>>>(V, uh, S);
    s4d_scan_local<<<S4D_NG, S4D_N, 0, stream>>>(zB, S, Gf);
    s4d_scan_groups<<<1, S4D_N, 0, stream>>>(zB, Gf, Cex);
    s4d_scan_apply<<<S4D_NG, S4D_N, 0, stream>>>(zB, S, Cex, Xh);
    s4d_out_gemm<<<(S4D_B / 16) * (S4D_M / 16), 32, 0, stream>>>(T, W, uh, Xh, out);
}
